// Model_65317862637884
// MI455X (gfx1250) — matrix operand order repaired and checked by reading, not compiled, not run on hardware
//
#include <hip/hip_runtime.h>
#include <hip/hip_bf16.h>

// ---------------------------------------------------------------------------
// FEDformer encoder forward on gfx1250 (MI455X).
// All heavy math is routed through v_wmma_f32_16x16x32_bf16 (f32 accumulate).
// GEMM uses LDS double buffering; transpose kernel uses async-to-LDS when the
// toolchain exposes the gfx1250 builtins (guarded, with plain fallback).
// ---------------------------------------------------------------------------

constexpr int BB  = 16;     // batch
constexpr int LL  = 1024;   // sequence length
constexpr int CIN = 64;     // input channels
constexpr int DD  = 512;    // model dim
constexpr int HH  = 8;      // heads
constexpr int EE  = 64;     // head dim
constexpr int DFF = 2048;   // FFN dim
constexpr int NLL = 2;      // layers
constexpr int MM  = 64;     // Fourier modes

typedef __attribute__((ext_vector_type(16))) __bf16 bf16x16;
typedef __attribute__((ext_vector_type(8)))  __bf16 bf16x8;
typedef __attribute__((ext_vector_type(8)))  float  f32x8;
typedef __attribute__((vector_size(16)))     int    i32x4;

union FragAB { bf16x16 v; bf16x8 h[2]; };

#if defined(__AMDGCN__) && \
    __has_builtin(__builtin_amdgcn_global_load_async_to_lds_b128) && \
    __has_builtin(__builtin_amdgcn_s_wait_asynccnt)
#define HAVE_ASYNC_LDS 1
#else
#define HAVE_ASYNC_LDS 0
#endif

// ---------------------------------------------------------------------------
// Generic batched GEMM:  C[z] = A[z] @ B[z]  (+bias) (+residual) (relu)
// A: [M x K] row-major (lda), B: [K x N] row-major (ldb), C: [M x N] (ldc).
// f32 in/out; tiles converted to bf16 while staging into LDS; WMMA 16x16x32
// bf16 with f32 accumulation. Block tile 128x128x32, 256 threads = 8 waves
// (2x4), wave tile 64x32. LDS double-buffered (one barrier per K step);
// next tile is fetched into registers during compute; K+2 tile prefetched.
// Requires: N % 128 == 0, K % 32 == 0 (M is guarded).
// ---------------------------------------------------------------------------
template<int BIAS, int RES, int RELU>
__global__ __launch_bounds__(256)
void gemm_bf16_k(const float* __restrict__ A, long lda, long sA,
                 const float* __restrict__ Bm, long ldb, long sB,
                 float* __restrict__ C, long ldc, long sC,
                 const float* __restrict__ bias,
                 const float* __restrict__ R, long ldr, long sR,
                 int M, int N, int K)
{
    constexpr int LDS_A = 40;   // 32 + 8 pad (bf16) -> 80B rows, 16B aligned
    constexpr int LDS_B = 40;
    __shared__ __align__(16) __bf16 As[2][128 * LDS_A];
    __shared__ __align__(16) __bf16 Bs[2][128 * LDS_B];

    const int z = blockIdx.z;
    A  += (size_t)z * sA;
    Bm += (size_t)z * sB;
    C  += (size_t)z * sC;
    if (RES) R += (size_t)z * sR;

    const int tid  = threadIdx.x;
    const int wave = tid >> 5, lane = tid & 31;
    const int lrow = lane & 15, hi = lane >> 4;
    const int wrr = wave >> 2, wcc = wave & 3;   // 2 x 4 wave grid
    const int mBase = blockIdx.y * 128;
    const int nBase = blockIdx.x * 128;
    const int wm = wrr * 64;
    const int wn = wcc * 32;

    const int ar = tid >> 3, ac = (tid & 7) << 2;    // A staging map (32 rows/pass)
    const int br = tid >> 5, bc = (tid & 31) << 2;   // B staging map (8 rows/pass)

    f32x8 acc[4][2];
#pragma unroll
    for (int i = 0; i < 4; ++i)
#pragma unroll
        for (int j = 0; j < 2; ++j) acc[i][j] = (f32x8)0.0f;

    float4 nA[4], nB[4];   // register staging for the next K tile

    auto loadG = [&](int k0) {
#pragma unroll
        for (int p = 0; p < 4; ++p) {
            const int grow = mBase + p * 32 + ar;
            nA[p] = make_float4(0.f, 0.f, 0.f, 0.f);
            if (grow < M) nA[p] = *(const float4*)(A + (size_t)grow * lda + k0 + ac);
            nB[p] = *(const float4*)(Bm + (size_t)(k0 + p * 8 + br) * ldb + nBase + bc);
        }
    };
    auto stage = [&](int buf) {   // f32 -> bf16; B transposed into [n][k]
#pragma unroll
        for (int p = 0; p < 4; ++p) {
            __bf16* da = &As[buf][(p * 32 + ar) * LDS_A + ac];
            da[0] = (__bf16)nA[p].x; da[1] = (__bf16)nA[p].y;
            da[2] = (__bf16)nA[p].z; da[3] = (__bf16)nA[p].w;
            const int krow = p * 8 + br;
            Bs[buf][(bc + 0) * LDS_B + krow] = (__bf16)nB[p].x;
            Bs[buf][(bc + 1) * LDS_B + krow] = (__bf16)nB[p].y;
            Bs[buf][(bc + 2) * LDS_B + krow] = (__bf16)nB[p].z;
            Bs[buf][(bc + 3) * LDS_B + krow] = (__bf16)nB[p].w;
        }
    };

    const int nK = K >> 5;
    loadG(0);
    stage(0);
    __syncthreads();

    for (int kb = 0; kb < nK; ++kb) {
        const int buf = kb & 1;
        const bool more = (kb + 1) < nK;
        if (more) loadG((kb + 1) << 5);
        if (kb + 2 < nK) {                       // L2 prefetch of K+2 tile
            const int k2 = (kb + 2) << 5;
            if (mBase + ar < M)
                __builtin_prefetch(A + (size_t)(mBase + ar) * lda + k2 + ac, 0, 1);
            __builtin_prefetch(Bm + (size_t)(k2 + br) * ldb + nBase + bc, 0, 1);
        }

        // ---- fragments per ISA 7.12.2 bf16 layouts ----
        FragAB af[4], bfr[2];
#pragma unroll
        for (int mt = 0; mt < 4; ++mt) {
            const __bf16* ap = &As[buf][(wm + mt * 16 + lrow) * LDS_A];
            af[mt].h[0] = *(const bf16x8*)(ap + hi * 8);       // K = hi*8 .. +7
            af[mt].h[1] = *(const bf16x8*)(ap + 16 + hi * 8);  // K = 16+hi*8 .. +7
        }
#pragma unroll
        for (int nt = 0; nt < 2; ++nt) {
            const __bf16* bp = &Bs[buf][(wn + nt * 16 + lrow) * LDS_B + hi * 8];
            bfr[nt].h[0] = *(const bf16x8*)(bp);               // K = hi*8 .. +7
            bfr[nt].h[1] = *(const bf16x8*)(bp + 16);          // K = 16+hi*8 .. +7
        }
#pragma unroll
        for (int mt = 0; mt < 4; ++mt)
#pragma unroll
            for (int nt = 0; nt < 2; ++nt)
                acc[mt][nt] = __builtin_amdgcn_wmma_f32_16x16x32_bf16(
                    false, af[mt].v, false, bfr[nt].v, (short)0, acc[mt][nt],
                    false, false);

        if (more) stage(buf ^ 1);   // write ping-pong buffer (safe: prev readers
                                    // of buf^1 finished before last barrier)
        __syncthreads();
    }

    // ---- epilogue: C layout m = 8*hi + r, n = lane%16 ----
#pragma unroll
    for (int mt = 0; mt < 4; ++mt)
#pragma unroll
        for (int nt = 0; nt < 2; ++nt) {
            const int col = nBase + wn + nt * 16 + lrow;
#pragma unroll
            for (int r8 = 0; r8 < 8; ++r8) {
                const int row = mBase + wm + mt * 16 + hi * 8 + r8;
                if (row < M) {
                    float v = acc[mt][nt][r8];
                    if (BIAS) v += bias[col];
                    if (RES)  v += R[(size_t)row * ldr + col];
                    if (RELU) v = fmaxf(v, 0.f);
                    C[(size_t)row * ldc + col] = v;
                }
            }
        }
}

// ---------------------------------------------------------------------------
// Fourier mode mixing: for each (h,m): complex [B x E] @ [E x E].
// X holds [b][3M][D] rows: 0..M-1 = xr, M..2M-1 = xi, 2M..3M-1 = -xi.
//   Gr = xr@Wr + (-xi)@Wi ,  Gi = xr@Wi + xi@Wr   (add-only accumulation,
// since bf16 WMMA has no A/B negate per the ISA NEG table).
// One 128-thread block (4 waves) per (h,m); wave w does o-columns [16w,16w+16).
// ---------------------------------------------------------------------------
__global__ __launch_bounds__(128)
void modemix_k(const float* __restrict__ X,
               const float* __restrict__ WrT, const float* __restrict__ WiT,
               float* __restrict__ G)
{
    __shared__ __align__(16) __bf16 Xs[3][16][72];   // xr / xi / -xi  [b][k]
    __shared__ __align__(16) __bf16 Ws[2][64][72];   // Wr / Wi        [o][i]

    const int hm = blockIdx.x;           // h*MM + m
    const int h  = hm >> 6;
    const int m  = hm & 63;
    const int tid = threadIdx.x;

    const size_t kindOff[3] = { (size_t)m * DD + (size_t)h * EE,
                                (size_t)(MM + m) * DD + (size_t)h * EE,
                                (size_t)(2 * MM + m) * DD + (size_t)h * EE };
#pragma unroll
    for (int kind = 0; kind < 3; ++kind)
#pragma unroll
        for (int p = 0; p < 2; ++p) {
            const int idx = p * 128 + tid;          // 0..255
            const int bb  = idx >> 4;
            const int cc  = (idx & 15) << 2;
            float4 v = *(const float4*)(X + (size_t)bb * (3 * MM * DD) +
                                        kindOff[kind] + cc);
            __bf16* d = &Xs[kind][bb][cc];
            d[0] = (__bf16)v.x; d[1] = (__bf16)v.y;
            d[2] = (__bf16)v.z; d[3] = (__bf16)v.w;
        }
    const float* Wsrc[2] = { WrT + (size_t)hm * EE * EE, WiT + (size_t)hm * EE * EE };
#pragma unroll
    for (int w2 = 0; w2 < 2; ++w2)
#pragma unroll
        for (int p = 0; p < 8; ++p) {
            const int idx = p * 128 + tid;          // 0..1023
            const int i = idx >> 4;
            const int o = (idx & 15) << 2;
            float4 v = *(const float4*)(Wsrc[w2] + (size_t)i * EE + o);
            Ws[w2][o + 0][i] = (__bf16)v.x;
            Ws[w2][o + 1][i] = (__bf16)v.y;
            Ws[w2][o + 2][i] = (__bf16)v.z;
            Ws[w2][o + 3][i] = (__bf16)v.w;
        }
    __syncthreads();

    const int wave = tid >> 5, lane = tid & 31;
    const int lrow = lane & 15, hi = lane >> 4;
    const int oBase = wave * 16;

    auto loadA = [&](int kind, int kc) -> bf16x16 {
        FragAB f;
        const __bf16* ap = &Xs[kind][lrow][0];
        f.h[0] = *(const bf16x8*)(ap + kc + hi * 8);
        f.h[1] = *(const bf16x8*)(ap + kc + 16 + hi * 8);
        return f.v;
    };
    auto loadB = [&](int w2, int kc) -> bf16x16 {
        FragAB f;
        const __bf16* bp = &Ws[w2][oBase + lrow][kc + hi * 8];
        f.h[0] = *(const bf16x8*)(bp);
        f.h[1] = *(const bf16x8*)(bp + 16);
        return f.v;
    };

    f32x8 aR = (f32x8)0.0f, aI = (f32x8)0.0f;
#pragma unroll
    for (int kc = 0; kc < 64; kc += 32) {
        bf16x16 xr  = loadA(0, kc);
        bf16x16 xi  = loadA(1, kc);
        bf16x16 xn  = loadA(2, kc);
        bf16x16 wrm = loadB(0, kc);
        bf16x16 wim = loadB(1, kc);
        aR = __builtin_amdgcn_wmma_f32_16x16x32_bf16(false, xr, false, wrm, (short)0, aR, false, false);
        aR = __builtin_amdgcn_wmma_f32_16x16x32_bf16(false, xn, false, wim, (short)0, aR, false, false);
        aI = __builtin_amdgcn_wmma_f32_16x16x32_bf16(false, xr, false, wim, (short)0, aI, false, false);
        aI = __builtin_amdgcn_wmma_f32_16x16x32_bf16(false, xi, false, wrm, (short)0, aI, false, false);
    }

    const int o = oBase + lrow;
#pragma unroll
    for (int r8 = 0; r8 < 8; ++r8) {
        const int bb = hi * 8 + r8;
        G[(size_t)bb * (2 * MM * DD) + (size_t)m * DD + h * EE + o]        = aR[r8];
        G[(size_t)bb * (2 * MM * DD) + (size_t)(MM + m) * DD + h * EE + o] = aI[r8];
    }
}

// ---------------------------------------------------------------------------
// Token embedding: circular conv1d(k=3) 64->512, +1e-10 on input, + pos-emb.
// ---------------------------------------------------------------------------
__global__ __launch_bounds__(256)
void embed_k(const float* __restrict__ x, const float* __restrict__ w,
             float* __restrict__ hOut)
{
    const int bl = blockIdx.x;
    const int b = bl >> 10, l = bl & 1023;
    __shared__ float xs[192];
    const int t = threadIdx.x;
    if (t < 192) {
        const int k = t / 64, c = t % 64;
        int ls = l + k - 1;
        if (ls < 0) ls += LL;
        if (ls >= LL) ls -= LL;
        xs[t] = x[((size_t)b * LL + ls) * CIN + c] + 1e-10f;
    }
    __syncthreads();
    float a0 = 0.f, a1 = 0.f;
    for (int i = 0; i < 192; ++i) {
        const float xv = xs[i];
        a0 += xv * w[(size_t)i * DD + t];
        a1 += xv * w[(size_t)i * DD + t + 256];
    }
    const float c1 = -9.210340371976184f / (float)DD;   // -ln(10000)/D
    auto pe = [&](int d) -> float {
        const int half = d >> 1;
        const float arg = (float)l * __expf((float)(2 * half) * c1);
        return (d & 1) ? __cosf(arg) : __sinf(arg);
    };
    const size_t base = (size_t)bl * DD;
    hOut[base + t]       = a0 + pe(t);
    hOut[base + t + 256] = a1 + pe(t + 256);
}

// ---------------------------------------------------------------------------
// DFT / iDFT basis generation (period-1024 angles reduced exactly in int).
// Fd rows: [0..63]=cos, [64..127]=-sin (xi), [128..191]=+sin (-xi trick).
// ---------------------------------------------------------------------------
__global__ void gen_dft_k(float* __restrict__ F)
{
    const int idx = blockIdx.x * 256 + threadIdx.x;
    if (idx >= 192 * 1024) return;
    const int row = idx >> 10, l = idx & 1023;
    const int m = row & 63, kind = row >> 6;
    const int r = (m * l) & 1023;
    const float ang = (6.283185307179586f / 1024.f) * (float)r;
    float v;
    if (kind == 0)      v =  __cosf(ang);
    else if (kind == 1) v = -__sinf(ang);
    else                v =  __sinf(ang);
    F[idx] = v;
}

__global__ void gen_irf_k(float* __restrict__ T)
{
    const int idx = blockIdx.x * 256 + threadIdx.x;
    if (idx >= 1024 * 128) return;
    const int l = idx >> 7, col = idx & 127;
    const int m = col & 63;
    const bool re = col < 64;
    float v;
    if (m == 0) {
        v = re ? (1.f / 1024.f) : 0.f;
    } else {
        const int r = (m * l) & 1023;
        const float ang = (6.283185307179586f / 1024.f) * (float)r;
        v = re ? (2.f / 1024.f) * __cosf(ang) : (-2.f / 1024.f) * __sinf(ang);
    }
    T[idx] = v;
}

// [NL,H,E,E,M] -> [NL,H,M,E,E]  (one-time, makes mode-mix B contiguous)
__global__ void wtrans_k(const float* __restrict__ in, float* __restrict__ out)
{
    const int idx = blockIdx.x * 256 + threadIdx.x;
    constexpr int TOT = NLL * HH * MM * EE * EE;
    if (idx >= TOT) return;
    const int o   = idx & 63;
    const int i   = (idx >> 6) & 63;
    const int m   = (idx >> 12) & 63;
    const int hh  = (idx >> 18) & 7;
    const int lyr = idx >> 21;
    const size_t src = ((((size_t)lyr * HH + hh) * EE + i) * EE + o) * MM + m;
    out[idx] = in[src];
}

// ---------------------------------------------------------------------------
// FEDformer reshape quirk: a[b, 2c+s, d] = a_nat[b, s*512+d, c].
// LDS-tiled 32x32 transpose; the global->LDS leg uses the gfx1250 async-to-LDS
// path (ASYNCcnt-tracked) when the builtin is available.
// ---------------------------------------------------------------------------
__global__ __launch_bounds__(256)
void scramble_k(const float* __restrict__ in, float* __restrict__ out)
{
    __shared__ __align__(16) float tile[32][36];
    const int b = blockIdx.z >> 1, s = blockIdx.z & 1;
    const float* ip = in  + (size_t)b * LL * DD + (size_t)s * 512 * DD;
    float*       op = out + (size_t)b * LL * DD + (size_t)s * DD;
    const int c0 = blockIdx.x * 32, d0 = blockIdx.y * 32;
    const int tid = threadIdx.x;

    const int row = tid >> 3;            // 0..31 (d within tile)
    const int c4  = (tid & 7) << 2;      // 0..28 (c within tile, x4)
    const float* gsrc = ip + (size_t)(d0 + row) * DD + c0 + c4;
#if HAVE_ASYNC_LDS
    __builtin_amdgcn_global_load_async_to_lds_b128(
        (__attribute__((address_space(1))) i32x4*)gsrc,
        (__attribute__((address_space(3))) i32x4*)&tile[row][c4],
        0, 0);
    __builtin_amdgcn_s_wait_asynccnt(0);
#else
    *(float4*)&tile[row][c4] = *(const float4*)gsrc;
#endif
    __syncthreads();

    const int c  = tid >> 3;             // 0..31 (c within tile)
    const int d4 = (tid & 7) << 2;       // 0..28 (d within tile, x4)
    float4 o;
    o.x = tile[d4 + 0][c];
    o.y = tile[d4 + 1][c];
    o.z = tile[d4 + 2][c];
    o.w = tile[d4 + 3][c];
    *(float4*)(op + (size_t)(c0 + c) * 2 * DD + d0 + d4) = o;
}

// y = x - moving_mean(x), window 25, replicate pad; sliding-window sums.
__global__ void movmean_k(const float* __restrict__ x, float* __restrict__ y)
{
    const int b = blockIdx.x >> 1;
    const int d = ((blockIdx.x & 1) << 8) + threadIdx.x;
    const int l0 = blockIdx.y << 7;               // 128-long chunks
    const float* p = x + (size_t)b * LL * DD + d;
    float*       o = y + (size_t)b * LL * DD + d;
    float s = 0.f;
    for (int j = l0 - 12; j <= l0 + 12; ++j) {
        int i = j < 0 ? 0 : (j > LL - 1 ? LL - 1 : j);
        s += p[(size_t)i * DD];
    }
    o[(size_t)l0 * DD] = p[(size_t)l0 * DD] - s * (1.f / 25.f);
    for (int l = l0 + 1; l < l0 + 128; ++l) {
        const int ia = (l + 12 > LL - 1) ? LL - 1 : l + 12;
        const int ir = (l - 13 < 0) ? 0 : l - 13;
        s += p[(size_t)ia * DD] - p[(size_t)ir * DD];
        o[(size_t)l * DD] = p[(size_t)l * DD] - s * (1.f / 25.f);
    }
}

// LayerNorm over D per (b,l).
__global__ __launch_bounds__(256)
void ln_k(const float* __restrict__ x, const float* __restrict__ g,
          const float* __restrict__ bt, float* __restrict__ y)
{
    const int bl = blockIdx.x;
    const float* xr = x + (size_t)bl * DD;
    const int t = threadIdx.x;
    const float v0 = xr[t], v1 = xr[t + 256];
    __shared__ float rs[256], rq[256];
    rs[t] = v0 + v1;
    rq[t] = v0 * v0 + v1 * v1;
    __syncthreads();
    for (int k = 128; k > 0; k >>= 1) {
        if (t < k) { rs[t] += rs[t + k]; rq[t] += rq[t + k]; }
        __syncthreads();
    }
    const float mu  = rs[0] * (1.f / DD);
    const float var = rq[0] * (1.f / DD) - mu * mu;
    const float inv = rsqrtf(var + 1e-5f);
    float* yr = y + (size_t)bl * DD;
    yr[t]       = (v0 - mu) * inv * g[t]       + bt[t];
    yr[t + 256] = (v1 - mu) * inv * g[t + 256] + bt[t + 256];
}

__global__ void colmean_k(const float* __restrict__ y, float* __restrict__ mc)
{
    const int b = blockIdx.x >> 1;
    const int d = ((blockIdx.x & 1) << 8) + threadIdx.x;
    const float* p = y + (size_t)b * LL * DD + d;
    float s = 0.f;
    for (int l = 0; l < LL; ++l) s += p[(size_t)l * DD];
    mc[b * DD + d] = s * (1.f / LL);
}

__global__ void final_k(const float* __restrict__ y, const float* __restrict__ mc,
                        float* __restrict__ out)
{
    const int b = blockIdx.x >> 1;
    const int d = ((blockIdx.x & 1) << 8) + threadIdx.x;
    const float* p = y + (size_t)b * LL * DD + d;
    const float m = mc[b * DD + d];
    float s = 0.f;
    for (int l = 0; l < LL; ++l) s += p[(size_t)l * DD] - m;
    out[b * DD + d] = s * (1.f / LL);
}

// ---------------------------------------------------------------------------
// Host side
// ---------------------------------------------------------------------------
template<int BIAS, int RES, int RELU>
static void launch_gemm(hipStream_t s,
                        const float* A, long lda, long sA,
                        const float* Bm, long ldb, long sB,
                        float* C, long ldc, long sC,
                        const float* bias,
                        const float* R, long ldr, long sR,
                        int M, int N, int K, int batch)
{
    dim3 grid(N / 128, (M + 127) / 128, batch), block(256);
    gemm_bf16_k<BIAS, RES, RELU><<<grid, block, 0, s>>>(
        A, lda, sA, Bm, ldb, sB, C, ldc, sC, bias, R, ldr, sR, M, N, K);
}

extern "C" void kernel_launch(void* const* d_in, const int* in_sizes, int n_in,
                              void* d_out, int out_size, void* d_ws, size_t ws_size,
                              hipStream_t stream)
{
    const float* x_enc  = (const float*)d_in[0];
    const float* emb_w  = (const float*)d_in[1];
    const float* Wq     = (const float*)d_in[2];
    const float* bq     = (const float*)d_in[3];
    const float* Wo     = (const float*)d_in[4];
    const float* bo     = (const float*)d_in[5];
    const float* four_r = (const float*)d_in[6];
    const float* four_i = (const float*)d_in[7];
    const float* W1     = (const float*)d_in[8];
    const float* W2     = (const float*)d_in[9];
    const float* ln_g   = (const float*)d_in[10];
    const float* ln_b   = (const float*)d_in[11];
    float* out = (float*)d_out;

    char* ws = (char*)d_ws;
    size_t off = 0;
    auto carve = [&](size_t bytes) -> float* {
        float* p = (float*)(ws + off);
        off += (bytes + 255) & ~(size_t)255;
        return p;
    };
    const size_t ACT = (size_t)BB * LL * DD * 4;
    float* S0  = carve(ACT);                                  // h
    float* S1  = carve(ACT);                                  // q / a_nat / x1 / z
    float* S2  = carve(ACT);                                  // scrambled attn out
    float* S3  = carve((size_t)BB * 3 * MM * DD * 4);         // xr|xi|-xi
    float* S4  = carve((size_t)BB * 2 * MM * DD * 4);         // Gr|Gi
    float* St  = carve((size_t)BB * LL * DFF * 4);            // FFN hidden
    float* WrT = carve((size_t)NLL * HH * MM * EE * EE * 4);
    float* WiT = carve((size_t)NLL * HH * MM * EE * EE * 4);
    float* Fd  = carve((size_t)192 * 1024 * 4);               // DFT basis [3M x L]
    float* Ti  = carve((size_t)1024 * 128 * 4);               // iDFT basis [L x 2M]
    float* Mb  = carve((size_t)BB * DD * 4);                  // column means
    (void)ws_size; (void)in_sizes; (void)n_in; (void)out_size;

    // per-call prep
    gen_dft_k<<<(192 * 1024 + 255) / 256, 256, 0, stream>>>(Fd);
    gen_irf_k<<<(1024 * 128 + 255) / 256, 256, 0, stream>>>(Ti);
    {
        const int TOT = NLL * HH * MM * EE * EE;
        wtrans_k<<<(TOT + 255) / 256, 256, 0, stream>>>(four_r, WrT);
        wtrans_k<<<(TOT + 255) / 256, 256, 0, stream>>>(four_i, WiT);
    }
    embed_k<<<BB * LL, 256, 0, stream>>>(x_enc, emb_w, S0);

    for (int l = 0; l < NLL; ++l) {
        const float* Wq_l = Wq + (size_t)l * DD * DD;
        const float* bq_l = bq + (size_t)l * DD;
        const float* Wo_l = Wo + (size_t)l * DD * DD;
        const float* bo_l = bo + (size_t)l * DD;
        const float* W1_l = W1 + (size_t)l * DD * DFF;
        const float* W2_l = W2 + (size_t)l * DFF * DD;
        const float* WrT_l = WrT + (size_t)l * HH * MM * EE * EE;
        const float* WiT_l = WiT + (size_t)l * HH * MM * EE * EE;

        // q = h @ Wq + bq                        [16384 x 512 x 512]
        launch_gemm<1, 0, 0>(stream, S0, DD, 0, Wq_l, DD, 0, S1, DD, 0,
                             bq_l, nullptr, 0, 0, BB * LL, DD, DD, 1);
        // X = Fd @ q_b  (rfft to 64 modes)       batched [192 x 512 x 1024]
        launch_gemm<0, 0, 0>(stream, Fd, LL, 0, S1, DD, (long)LL * DD,
                             S3, DD, (long)3 * MM * DD,
                             nullptr, nullptr, 0, 0, 3 * MM, DD, LL, BB);
        // complex mode mixing
        modemix_k<<<HH * MM, 128, 0, stream>>>(S3, WrT_l, WiT_l, S4);
        // a_nat = Ti @ G_b (irfft)               batched [1024 x 512 x 128]
        launch_gemm<0, 0, 0>(stream, Ti, 2 * MM, 0, S4, DD, (long)2 * MM * DD,
                             S1, DD, (long)LL * DD,
                             nullptr, nullptr, 0, 0, LL, DD, 2 * MM, BB);
        // FEDformer view() scramble
        scramble_k<<<dim3(16, 16, BB * 2), 256, 0, stream>>>(S1, S2);
        // x1 = h + a @ Wo + bo                   [16384 x 512 x 512]
        launch_gemm<1, 1, 0>(stream, S2, DD, 0, Wo_l, DD, 0, S1, DD, 0,
                             bo_l, S0, DD, 0, BB * LL, DD, DD, 1);
        // decomp1
        movmean_k<<<dim3(BB * 2, 8), 256, 0, stream>>>(S1, S0);
        // t = relu(x @ W1)                       [16384 x 2048 x 512]
        launch_gemm<0, 0, 1>(stream, S0, DD, 0, W1_l, DFF, 0, St, DFF, 0,
                             nullptr, nullptr, 0, 0, BB * LL, DFF, DD, 1);
        // z = x + t @ W2                         [16384 x 512 x 2048]
        launch_gemm<0, 1, 0>(stream, St, DFF, 0, W2_l, DD, 0, S1, DD, 0,
                             nullptr, S0, DD, 0, BB * LL, DD, DFF, 1);
        // decomp2 -> h for next layer
        movmean_k<<<dim3(BB * 2, 8), 256, 0, stream>>>(S1, S0);
    }

    // my_Layernorm + mean over L
    ln_k<<<BB * LL, 256, 0, stream>>>(S0, ln_g, ln_b, S1);
    colmean_k<<<BB * 2, 256, 0, stream>>>(S1, Mb);
    final_k<<<BB * 2, 256, 0, stream>>>(S1, Mb, out);
}
